// JointNet_11931419148510
// MI455X (gfx1250) — matrix operand order repaired and checked by reading, not compiled, not run on hardware
//
#include <hip/hip_runtime.h>
#include <hip/hip_bf16.h>

// ---------- types ----------
typedef __bf16        v16bf __attribute__((ext_vector_type(16)));
typedef float         v8f   __attribute__((ext_vector_type(8)));
typedef float         v4f   __attribute__((ext_vector_type(4)));
typedef unsigned int  v4u   __attribute__((ext_vector_type(4)));

union FragBF { v4u u[2]; v16bf v; };

// float -> bf16, round-to-nearest-even
__device__ __forceinline__ unsigned short f2bf(float f) {
    unsigned int u = __float_as_uint(f);
    u += 0x7FFFu + ((u >> 16) & 1u);
    return (unsigned short)(u >> 16);
}

// ---------- small fp32 projection: y[row][j] = sum_d x[row][d]*W[j][d] (+bias[j]) ----------
// x: (R,512), W: (512,512) row-major (J,D), y: (R,512). One block per row.
__global__ __launch_bounds__(256) void lin_kernel(const float* __restrict__ x,
                                                  const float* __restrict__ W,
                                                  const float* __restrict__ bias,
                                                  float* __restrict__ y) {
    __shared__ float xs[512];
    const int row = blockIdx.x;
    const int tid = threadIdx.x;
    xs[tid]       = x[(size_t)row * 512 + tid];
    xs[tid + 256] = x[(size_t)row * 512 + tid + 256];
    __syncthreads();
    for (int j = tid; j < 512; j += 256) {
        const float* wr = W + (size_t)j * 512;
        float acc = bias ? bias[j] : 0.0f;
        #pragma unroll 4
        for (int d = 0; d < 512; d += 4) {
            v4f w = *(const v4f*)(wr + d);
            acc += xs[d]     * w.x;
            acc += xs[d + 1] * w.y;
            acc += xs[d + 2] * w.z;
            acc += xs[d + 3] * w.w;
        }
        y[(size_t)row * 512 + j] = acc;
    }
}

// ---------- fp32 -> bf16 conversion (for W_out) ----------
__global__ __launch_bounds__(256) void cvt_bf16(const float* __restrict__ in,
                                                unsigned short* __restrict__ out, int n) {
    int i = blockIdx.x * 256 + threadIdx.x;
    if (i < n) out[i] = f2bf(in[i]);
}

// ---------- fused tanh + output GEMM via bf16 WMMA ----------
// grid: (625 row-tiles of 16 within T*U=10000, 8 batches). 256 threads = 8 waves.
// Each wave: 16 rows x 128 cols (8 n-tiles of 16), K=512 in chunks of 32.
#define LDS_STRIDE 520  // ushorts per z-tile row (512 + 8 pad -> bank-conflict-free A loads)

__global__ __launch_bounds__(256) void joint_main(const float* __restrict__ e,           // (1600,512)
                                                  const float* __restrict__ dpre,        // (400,512)
                                                  const unsigned short* __restrict__ Wo, // (1024,512) bf16
                                                  const float* __restrict__ b_out,       // (1024)
                                                  float* __restrict__ out) {             // (8,200,50,1024)
    __shared__ unsigned short zt[16 * LDS_STRIDE];

    const int rt      = blockIdx.x;      // row tile: 0..624
    const int b       = blockIdx.y;      // batch:    0..7
    const int rowBase = rt * 16;         // row index within batch's (T*U)
    const int tid     = threadIdx.x;

    // ---- build z tile: z[r][j] = tanh(e[b*T + t][j] + d[b*U + u][j]) in bf16 ----
    for (int i = tid; i < 16 * 512; i += 256) {      // exactly 32 iterations, no divergence
        const int r = i >> 9;
        const int j = i & 511;
        const int g = rowBase + r;
        const int t = g / 50;
        const int u = g - t * 50;
        const float ev = e[((size_t)b * 200 + t) * 512 + j];
        const float dv = dpre[((size_t)b * 50 + u) * 512 + j];
        zt[r * LDS_STRIDE + j] = f2bf(tanhf(ev + dv));
    }
    __syncthreads();

    const int lane    = tid & 31;
    const int wave    = tid >> 5;
    const int m       = lane & 15;   // A row / B,D column-within-tile
    const int h       = lane >> 4;   // lane half
    const int colBase = wave * 128;  // this wave's V-column slab

    v8f acc[8] = {};

    for (int kk = 0; kk < 512; kk += 32) {
        // A fragment (16x32 bf16), ISA lane layout:
        //   vgprs 0..3 : k = kk + h*8 + {0..7};  vgprs 4..7 : k = kk + 16 + h*8 + {0..7}
        FragBF a;
        a.u[0] = *(const v4u*)&zt[m * LDS_STRIDE + kk + h * 8];
        a.u[1] = *(const v4u*)&zt[m * LDS_STRIDE + kk + 16 + h * 8];

        #pragma unroll
        for (int nt = 0; nt < 8; ++nt) {
            // B fragment (32x16 bf16): lane holds col n, k = kk + h*8 + {0..7}, kk + 16 + h*8 + {0..7}
            const int n = colBase + nt * 16 + m;
            const unsigned short* wp = Wo + (size_t)n * 512 + kk + h * 8;
            FragBF bm;
            bm.u[0] = *(const v4u*)(wp);
            bm.u[1] = *(const v4u*)(wp + 16);

            acc[nt] = __builtin_amdgcn_wmma_f32_16x16x32_bf16(
                false, a.v, false, bm.v, (short)0, acc[nt], false, false);
        }
    }

    // ---- epilogue: D vgpr r, lane -> (M = r + 8*h, N = m); add b_out, store fp32 ----
    #pragma unroll
    for (int nt = 0; nt < 8; ++nt) {
        const int   ncol = colBase + nt * 16 + m;
        const float bo   = b_out[ncol];
        #pragma unroll
        for (int r = 0; r < 8; ++r) {
            const int M = r + 8 * h;
            const int g = rowBase + M;
            const int t = g / 50;
            const int u = g - t * 50;
            out[(((size_t)b * 200 + t) * 50 + u) * 1024 + ncol] = acc[nt][r] + bo;
        }
    }
}

// ---------- launcher ----------
extern "C" void kernel_launch(void* const* d_in, const int* in_sizes, int n_in,
                              void* d_out, int out_size, void* d_ws, size_t ws_size,
                              hipStream_t stream) {
    const float* h_enc = (const float*)d_in[0];  // (8,200,1,512)
    const float* h_dec = (const float*)d_in[1];  // (8,1,50,512)
    const float* W_enc = (const float*)d_in[2];  // (512,512)
    const float* b_enc = (const float*)d_in[3];  // (512)
    const float* W_dec = (const float*)d_in[4];  // (512,512)
    const float* W_out = (const float*)d_in[5];  // (1024,512)
    const float* b_out = (const float*)d_in[6];  // (1024)
    float* out = (float*)d_out;                  // (8,200,50,1024)

    // workspace layout
    char* ws = (char*)d_ws;
    float* e_ws = (float*)ws;                                 // 1600*512 fp32 = 3.125 MB
    ws += (size_t)1600 * 512 * sizeof(float);
    float* d_ws_proj = (float*)ws;                            // 400*512 fp32 = 0.78 MB
    ws += (size_t)400 * 512 * sizeof(float);
    unsigned short* Wo_bf = (unsigned short*)ws;              // 1024*512 bf16 = 1 MB

    // 1) e = h_enc @ W_enc^T + b_enc   (1600 rows)
    lin_kernel<<<1600, 256, 0, stream>>>(h_enc, W_enc, b_enc, e_ws);
    // 2) d = h_dec @ W_dec^T           (400 rows, no bias)
    lin_kernel<<<400, 256, 0, stream>>>(h_dec, W_dec, nullptr, d_ws_proj);
    // 3) W_out -> bf16
    cvt_bf16<<<(1024 * 512) / 256, 256, 0, stream>>>(W_out, Wo_bf, 1024 * 512);
    // 4) fused tanh + output GEMM (WMMA bf16)
    joint_main<<<dim3(625, 8), 256, 0, stream>>>(e_ws, d_ws_proj, Wo_bf, b_out, out);
}
